// HypergraphConvLayer_78847009620372
// MI455X (gfx1250) — matrix operand order repaired and checked by reading, not compiled, not run on hardware
//
#include <hip/hip_runtime.h>
#include <hip/hip_bf16.h>

// Problem constants (match reference)
#define NN 12288
#define EE 6144
#define CC 256

#define KSTEP 64
#define LDPAD 66   // u16 row stride for LDS tile (even; padded vs 64 to dodge bank conflicts)

typedef __attribute__((ext_vector_type(16))) __bf16 v16bf;
typedef __attribute__((ext_vector_type(8)))  float  v8f;

union BF16Frag { unsigned u[8]; v16bf v; };

// float -> bf16 bits, round-to-nearest-even
__device__ __forceinline__ unsigned f2bf_bits(float f) {
    unsigned x = __float_as_uint(f);
    return (x + 0x7FFFu + ((x >> 16) & 1u)) >> 16;
}
__device__ __forceinline__ unsigned short f2bf16(float f) {
    return (unsigned short)f2bf_bits(f);
}

// B-fragment (32x16 bf16): 16 contiguous bf16 per lane = two b128 loads.
// Lanes 0-15 hold K=0..15 (2 per VGPR), lanes 16-31 hold K=16..31.
__device__ __forceinline__ BF16Frag load_bfrag_at(const unsigned short* __restrict__ p) {
    BF16Frag f;
    uint4 a = *(const uint4*)p;
    uint4 b = *(const uint4*)(p + 8);
    f.u[0] = a.x; f.u[1] = a.y; f.u[2] = a.z; f.u[3] = a.w;
    f.u[4] = b.x; f.u[5] = b.y; f.u[6] = b.z; f.u[7] = b.w;
    return f;
}

// A-fragment (16x32 bf16) from LDS tile lds16[64][LDPAD] at K offset kbase.
// A layout (contiguous K, matches B): lanes 0-15 (h=0): K in 0..15; lanes 16-31 (h=1): K in 16..31;
// each VGPR packs (K, K+1), K even -> aligned u32 reads (merge to ds_load_2addr).
__device__ __forceinline__ BF16Frag load_afrag_lds(const unsigned short* __restrict__ lds16,
                                                   int m, int kbase, int h) {
    BF16Frag f;
    const unsigned* row = (const unsigned*)(lds16 + m * LDPAD);
#pragma unroll
    for (int v = 0; v < 8; ++v) {
        int k = kbase + 16 * h + 2 * v;
        f.u[v] = row[k >> 1];
    }
    return f;
}

// ---------------------------------------------------------------------------
// One-shot: Wt[j][k] = bf16(W[k][j])   (transposed bf16 panel, 128 KB)
// ---------------------------------------------------------------------------
__global__ void __launch_bounds__(256)
convert_w_kernel(const float* __restrict__ W, unsigned short* __restrict__ Wt) {
    int j  = blockIdx.x * 64 + (threadIdx.x & 63);
    int k0 = (threadIdx.x >> 6) * 64;
#pragma unroll 4
    for (int kk = 0; kk < 64; ++kk) {
        int k = k0 + kk;
        Wt[(size_t)j * CC + k] = f2bf16(W[(size_t)k * CC + j]);
    }
}

// ---------------------------------------------------------------------------
// Generic WMMA GEMM:  out = Ddiag_row_scale * (op(A) @ Bt^T [+ bias])
//   TRANS_A=1 : opA[m][k] = A[k*LDA + m]   (H^T path, m over E)
//   TRANS_A=0 : opA[m][k] = A[m*LDA + k]   (H and x paths)
//   OUT_BF16T : store bf16 column-major [CC][MDIM] panel, else fp32 row-major
// All dims are template constants so per-tile displacements fold into the
// 24-bit instruction offsets (one address VGPR pair for all 8 B fragments).
// Block: 256 threads = 8 waves; 64-row M tile x all 256 columns.
// ---------------------------------------------------------------------------
template <bool TRANS_A, bool OUT_BF16T, bool HAS_BIAS, int LDA, int KDIM, int MDIM, int DDIAGN>
__global__ void __launch_bounds__(256)
hgemm_kernel(const float* __restrict__ A,
             const unsigned short* __restrict__ Bt,
             const float* __restrict__ Ddiag,
             const float* __restrict__ bias,
             unsigned short* __restrict__ outBT,
             float* __restrict__ outF) {
    __shared__ unsigned short lds16[64 * LDPAD];
    const int tid  = threadIdx.x;
    const int wave = tid >> 5;
    const int lane = tid & 31;
    const int jl   = lane & 15;
    const int h    = lane >> 4;
    const int m0   = blockIdx.x * 64;
    const int msub = wave >> 1;           // 0..3 -> 16-row subtile
    const int jbase = (wave & 1) * 128;   // 8 column subtiles of 16

    // One base pointer for all B fragments; t/ks displacements are constexpr.
    const unsigned short* bbase =
        Bt + (size_t)(jbase + jl) * KDIM + 16 * h;

    // Staging base pointers (per-thread), advanced by constexpr strides.
    const int sm = tid & 63;              // TRANS_A: m_loc ; else k_loc
    const int sg = tid >> 6;              // 0..3
    const float* abase;
    if (TRANS_A) abase = A + (size_t)sg * 16 * LDA + (m0 + sm);
    else         abase = A + (size_t)(m0 + sg) * LDA + sm;

    v8f acc[8] = {};

    for (int k0 = 0; k0 < KDIM; k0 += KSTEP) {
        __syncthreads();
        if (TRANS_A) {
            // A[k0..k0+63][m0..m0+63], coalesced along m (contiguous H columns)
            const float* src = abase + (size_t)k0 * LDA;
#pragma unroll
            for (int kk = 0; kk < 16; ++kk) {
                float v = __builtin_nontemporal_load(src + (size_t)kk * LDA);
                lds16[sm * LDPAD + sg * 16 + kk] = f2bf16(v);
            }
            if (k0 + KSTEP < KDIM) {
#pragma unroll
                for (int kk = 0; kk < 16; kk += 4)
                    __builtin_prefetch(src + (size_t)(KSTEP + kk) * LDA, 0, 0);
            }
        } else {
            // A[m0..m0+63][k0..k0+63], coalesced along k
            const float* src = abase + k0;
#pragma unroll
            for (int mm = 0; mm < 16; ++mm) {
                float v = __builtin_nontemporal_load(src + (size_t)mm * 4 * LDA);
                lds16[(sg + 4 * mm) * LDPAD + sm] = f2bf16(v);
            }
            if (k0 + KSTEP < KDIM) {
#pragma unroll
                for (int mm = 0; mm < 16; mm += 4)
                    __builtin_prefetch(src + (size_t)mm * 4 * LDA + KSTEP, 0, 0);
            }
        }
        __syncthreads();

#pragma unroll
        for (int ks = 0; ks < KSTEP; ks += 32) {
            BF16Frag afrag = load_afrag_lds(lds16, msub * 16 + jl, ks, h);
            const unsigned short* bk = bbase + k0 + ks;
#pragma unroll
            for (int t = 0; t < 8; ++t) {
                BF16Frag bfrag = load_bfrag_at(bk + (size_t)t * 16 * KDIM);
                acc[t] = __builtin_amdgcn_wmma_f32_16x16x32_bf16(
                    false, afrag.v, false, bfrag.v, (short)0, acc[t], false, false);
            }
        }
    }

    // epilogue: row scale from diagonal of dense D matrix [+ bias]
    float scl[8];
#pragma unroll
    for (int r = 0; r < 8; ++r) {
        int m = m0 + msub * 16 + r + 8 * h;
        scl[r] = Ddiag[(size_t)m * (size_t)(DDIAGN + 1)];
    }
#pragma unroll
    for (int t = 0; t < 8; ++t) {
        int j = jbase + t * 16 + jl;
        float add = HAS_BIAS ? bias[j] : 0.0f;
#pragma unroll
        for (int r = 0; r < 8; ++r) {
            int m = m0 + msub * 16 + r + 8 * h;
            float val = scl[r] * (acc[t][r] + add);
            if (OUT_BF16T) {
                outBT[(size_t)j * MDIM + m] = f2bf16(val);
            } else {
                __builtin_nontemporal_store(val, outF + (size_t)m * CC + j);
            }
        }
    }
}

extern "C" void kernel_launch(void* const* d_in, const int* in_sizes, int n_in,
                              void* d_out, int out_size, void* d_ws, size_t ws_size,
                              hipStream_t stream) {
    const float* x  = (const float*)d_in[0];   // [N, 256]
    const float* H  = (const float*)d_in[1];   // [N, E]
    const float* Dv = (const float*)d_in[2];   // [N, N] dense diag
    const float* De = (const float*)d_in[3];   // [E, E] dense diag
    const float* W  = (const float*)d_in[4];   // [256, 256]
    const float* b  = (const float*)d_in[5];   // [256]
    float* out = (float*)d_out;                // [N, 256]

    // Workspace: Yt bf16 [256][N], Zt bf16 [256][E], Wt bf16 [256][256]  (~9.6 MB)
    unsigned short* Yt = (unsigned short*)d_ws;
    unsigned short* Zt = Yt + (size_t)CC * NN;
    unsigned short* Wt = Zt + (size_t)CC * EE;

    dim3 blk(256);
    // W^T -> bf16 panel (once)
    convert_w_kernel<<<4, blk, 0, stream>>>(W, Wt);
    // Stage A: Yt = (Dv * (x @ W + b))^T   (bf16, column-major)
    hgemm_kernel<false, true, true, CC, CC, NN, NN><<<NN / 64, blk, 0, stream>>>(
        x, Wt, Dv, b, Yt, nullptr);
    // Stage B: Zt = (De * (H^T @ Y))^T     (bf16, column-major)
    hgemm_kernel<true, true, false, EE, NN, EE, EE><<<EE / 64, blk, 0, stream>>>(
        H, Yt, De, nullptr, Zt, nullptr);
    // Stage C: out = Dv * (H @ Z)          (fp32, row-major, nontemporal)
    hgemm_kernel<false, false, false, EE, EE, NN, NN><<<NN / 64, blk, 0, stream>>>(
        H, Zt, Dv, nullptr, nullptr, out);
}
